// GNN_78288663872079
// MI455X (gfx1250) — compile-verified, matrix operand order checked by reading, not run on hardware
//
#include <hip/hip_runtime.h>
#include <hip/hip_bf16.h>

typedef __attribute__((ext_vector_type(16))) _Float16 v16h;
typedef __attribute__((ext_vector_type(8)))  _Float16 v8h;
typedef __attribute__((ext_vector_type(8)))  float    v8f;

union U16 { v16h v; v8h h[2]; };

#define N_NODES 50000
#define N_EDGES 800000
#define D_IN    128
#define H1      512
#define H2      256
#define H3      256
#define N_CLASS 10
#define N_GRAPHS 256

// ---------------------------------------------------------------------------
// float4 copy (init aggregation buffer with self term)
__global__ void copy4_kernel(float* __restrict__ dst, const float* __restrict__ src, long long n4) {
    long long i = (long long)blockIdx.x * blockDim.x + threadIdx.x;
    long long stride = (long long)gridDim.x * blockDim.x;
    for (; i < n4; i += stride)
        ((float4*)dst)[i] = ((const float4*)src)[i];
}

// zero-fill
__global__ void zero_kernel(float* __restrict__ p, long long n) {
    long long i = (long long)blockIdx.x * blockDim.x + threadIdx.x;
    long long stride = (long long)gridDim.x * blockDim.x;
    for (; i < n; i += stride) p[i] = 0.0f;
}

// f32 -> f16 convert (contiguous)
__global__ void cvt_f16_kernel(_Float16* __restrict__ dst, const float* __restrict__ src, long long n) {
    long long i = (long long)blockIdx.x * blockDim.x + threadIdx.x;
    long long stride = (long long)gridDim.x * blockDim.x;
    for (; i < n; i += stride) dst[i] = (_Float16)src[i];
}

// W[K][N] f32 -> Wt[N][K] f16 (so B fragments load contiguous K per lane)
__global__ void transconv_kernel(_Float16* __restrict__ wt, const float* __restrict__ w, int K, int N) {
    long long total = (long long)K * N;
    long long i = (long long)blockIdx.x * blockDim.x + threadIdx.x;
    long long stride = (long long)gridDim.x * blockDim.x;
    for (; i < total; i += stride) {
        int k = (int)(i / N);
        int n = (int)(i % N);
        wt[(size_t)n * K + k] = (_Float16)w[i];
    }
}

// ---------------------------------------------------------------------------
// Edge scatter-add: agg[dst] += x[src], float4 gather + 4x global_atomic_add_f32.
// d4shift: log2(D/4). Targets fit in 192MB L2 -> atomics mostly L2-resident.
__global__ void edge_scatter_kernel(const float* __restrict__ x, float* __restrict__ agg,
                                    const int* __restrict__ src, const int* __restrict__ dst,
                                    long long total, int d4shift) {
    long long i = (long long)blockIdx.x * blockDim.x + threadIdx.x;
    long long stride = (long long)gridDim.x * blockDim.x;
    const int mask = (1 << d4shift) - 1;
    const int D = (mask + 1) << 2;
    for (; i < total; i += stride) {
        int e = (int)(i >> d4shift);
        int c = ((int)i & mask) << 2;
        float4 v = *(const float4*)(x + (size_t)src[e] * D + c);
        float* p = agg + (size_t)dst[e] * D + c;
        unsafeAtomicAdd(p + 0, v.x);
        unsafeAtomicAdd(p + 1, v.y);
        unsafeAtomicAdd(p + 2, v.z);
        unsafeAtomicAdd(p + 3, v.w);
    }
}

// ---------------------------------------------------------------------------
// WMMA GEMM: C[M][N] = act(A[M][K](f16) @ Bt[N][K](f16)^T + bias), f32 accumulate.
// Block = 256 threads = 8 waves; each wave does a 16(M) x 64(N) tile:
// 4x v_wmma_f32_16x16x32_f16 per K-step, reusing one A fragment.
// A-fragment layout per ISA 7.12.2 (16-bit A 16x32): lane&15 = row,
// lane>>4 selects K half-group; elements 0..7 at K+kg*8, 8..15 at K+16+kg*8.
__global__ __launch_bounds__(256) void gemm_wmma_kernel(
    const _Float16* __restrict__ A, const _Float16* __restrict__ Bt,
    const float* __restrict__ bias, float* __restrict__ C,
    int M, int K, int N, int relu)
{
    const int lane = threadIdx.x & 31;
    const int wave = threadIdx.x >> 5;
    const int row0 = blockIdx.y * 128 + wave * 16;
    if (row0 >= M) return;               // wave-uniform: EXEC stays all-ones
    const int col0 = blockIdx.x * 64;
    const int r  = lane & 15;
    const int kg = lane >> 4;

    const _Float16* Arow = A  + (size_t)(row0 + r) * K + (kg << 3);
    const _Float16* Brow = Bt + (size_t)(col0 + r) * K + (kg << 3);
    const size_t bstep = (size_t)16 * K;

    v8f acc[4] = {v8f{}, v8f{}, v8f{}, v8f{}};

    for (int k0 = 0; k0 < K; k0 += 32) {
        U16 a;
        a.h[0] = *(const v8h*)(Arow + k0);
        a.h[1] = *(const v8h*)(Arow + k0 + 16);
#pragma unroll
        for (int s = 0; s < 4; ++s) {
            U16 b;
            const _Float16* bp = Brow + (size_t)s * bstep;
            b.h[0] = *(const v8h*)(bp + k0);
            b.h[1] = *(const v8h*)(bp + k0 + 16);
            acc[s] = __builtin_amdgcn_wmma_f32_16x16x32_f16(
                false, a.v, false, b.v, (short)0, acc[s], false, false);
        }
    }

    // C layout (32-bit 16x16): VGPR j -> M = j + 8*kg, column = lane&15
#pragma unroll
    for (int s = 0; s < 4; ++s) {
        const int col = col0 + s * 16 + r;
        const float bv = bias[col];
#pragma unroll
        for (int j = 0; j < 8; ++j) {
            float v = acc[s][j] + bv;
            if (relu) v = fmaxf(v, 0.0f);
            C[(size_t)(row0 + (kg << 3) + j) * N + col] = v;
        }
    }
}

// ---------------------------------------------------------------------------
// Graph pooling: pooled[idx[n]][c] += x2[n][c]
__global__ void pool_kernel(const float* __restrict__ x2, const int* __restrict__ idx,
                            float* __restrict__ pooled, long long total, int hshift) {
    long long i = (long long)blockIdx.x * blockDim.x + threadIdx.x;
    long long stride = (long long)gridDim.x * blockDim.x;
    const int mask = (1 << hshift) - 1;
    for (; i < total; i += stride) {
        int n = (int)(i >> hshift);
        int c = (int)i & mask;
        unsafeAtomicAdd(&pooled[((size_t)idx[n] << hshift) + c], x2[i]);
    }
}

// ---------------------------------------------------------------------------
// Head: relu(pooled @ W3 + b3) @ W4 + b4 -> log_softmax. One block per graph.
__global__ __launch_bounds__(256) void head_kernel(
    const float* __restrict__ pooled,
    const float* __restrict__ W3, const float* __restrict__ b3,
    const float* __restrict__ W4, const float* __restrict__ b4,
    float* __restrict__ out)
{
    __shared__ float prow[H2];
    __shared__ float hrow[H3];
    __shared__ float o[N_CLASS];
    __shared__ float red[2];
    const int g = blockIdx.x;
    const int t = threadIdx.x;

    prow[t] = pooled[(size_t)g * H2 + t];
    __syncthreads();

    float acc = b3[t];
    for (int k = 0; k < H2; ++k) acc += prow[k] * W3[(size_t)k * H3 + t];
    hrow[t] = fmaxf(acc, 0.0f);
    __syncthreads();

    if (t < N_CLASS) {
        float a = b4[t];
        for (int k = 0; k < H3; ++k) a += hrow[k] * W4[(size_t)k * N_CLASS + t];
        o[t] = a;
    }
    __syncthreads();

    if (t == 0) {
        float m = o[0];
        for (int i = 1; i < N_CLASS; ++i) m = fmaxf(m, o[i]);
        float s = 0.0f;
        for (int i = 0; i < N_CLASS; ++i) s += expf(o[i] - m);
        red[0] = m;
        red[1] = logf(s);
    }
    __syncthreads();

    if (t < N_CLASS) out[(size_t)g * N_CLASS + t] = o[t] - red[0] - red[1];
}

// ---------------------------------------------------------------------------
static inline int gsBlocks(long long total) {
    long long b = (total + 255) / 256;
    if (b > 262144) b = 262144;
    if (b < 1) b = 1;
    return (int)b;
}

extern "C" void kernel_launch(void* const* d_in, const int* in_sizes, int n_in,
                              void* d_out, int out_size, void* d_ws, size_t ws_size,
                              hipStream_t stream) {
    const float* x_in = (const float*)d_in[0];
    const int*   esrc = (const int*)d_in[1];
    const int*   edst = (const int*)d_in[2];
    const int*   idx  = (const int*)d_in[3];
    const float* W1 = (const float*)d_in[4];  const float* b1 = (const float*)d_in[5];
    const float* W2 = (const float*)d_in[6];  const float* b2 = (const float*)d_in[7];
    const float* W3 = (const float*)d_in[8];  const float* b3 = (const float*)d_in[9];
    const float* W4 = (const float*)d_in[10]; const float* b4 = (const float*)d_in[11];
    float* out = (float*)d_out;

    // Workspace carve-up (aliased across pipeline stages, all stream-ordered)
    char* ws = (char*)d_ws;
    size_t off = 0;
    auto carve = [&](size_t bytes) -> char* {
        char* p = ws + off;
        off += (bytes + 255) & ~(size_t)255;
        return p;
    };
    float*    R1     = (float*)   carve((size_t)N_NODES * H1 * 4);   // z1, then x2
    float*    R2     = (float*)   carve((size_t)N_NODES * H1 * 4);   // agg1, then agg2 (f32)
    _Float16* a1h    = (_Float16*)carve((size_t)N_NODES * D_IN * 2);
    _Float16* a2h    = (_Float16*)carve((size_t)N_NODES * H1 * 2);
    _Float16* w1t    = (_Float16*)carve((size_t)H1 * D_IN * 2);      // [H1][D_IN]
    _Float16* w2t    = (_Float16*)carve((size_t)H2 * H1 * 2);        // [H2][H1]
    float*    pooled = (float*)   carve((size_t)N_GRAPHS * H2 * 4);

    // --- Stage 0: weight prep (f32 -> f16, transposed) ---
    transconv_kernel<<<gsBlocks((long long)D_IN * H1), 256, 0, stream>>>(w1t, W1, D_IN, H1);
    transconv_kernel<<<gsBlocks((long long)H1 * H2),   256, 0, stream>>>(w2t, W2, H1, H2);

    // --- Stage 1: agg1 = x + scatter_add(x[src] -> dst) ---
    copy4_kernel<<<gsBlocks((long long)N_NODES * D_IN / 4), 256, 0, stream>>>(
        R2, x_in, (long long)N_NODES * D_IN / 4);
    edge_scatter_kernel<<<8192, 256, 0, stream>>>(
        x_in, R2, esrc, edst, (long long)N_EDGES * (D_IN / 4), 5 /*log2(128/4)*/);
    cvt_f16_kernel<<<gsBlocks((long long)N_NODES * D_IN), 256, 0, stream>>>(
        a1h, R2, (long long)N_NODES * D_IN);

    // --- Stage 2: z1 = relu(agg1 @ W1 + b1)  [WMMA] ---
    {
        dim3 grid(H1 / 64, (N_NODES + 127) / 128);
        gemm_wmma_kernel<<<grid, 256, 0, stream>>>(a1h, w1t, b1, R1,
                                                   N_NODES, D_IN, H1, 1);
    }

    // --- Stage 3: agg2 = z1 + scatter_add(z1[src] -> dst) ---
    copy4_kernel<<<gsBlocks((long long)N_NODES * H1 / 4), 256, 0, stream>>>(
        R2, R1, (long long)N_NODES * H1 / 4);
    edge_scatter_kernel<<<16384, 256, 0, stream>>>(
        R1, R2, esrc, edst, (long long)N_EDGES * (H1 / 4), 7 /*log2(512/4)*/);
    cvt_f16_kernel<<<gsBlocks((long long)N_NODES * H1), 256, 0, stream>>>(
        a2h, R2, (long long)N_NODES * H1);

    // --- Stage 4: x2 = agg2 @ W2 + b2  [WMMA] (overwrites R1; z1 is dead) ---
    {
        dim3 grid(H2 / 64, (N_NODES + 127) / 128);
        gemm_wmma_kernel<<<grid, 256, 0, stream>>>(a2h, w2t, b2, R1,
                                                   N_NODES, H1, H2, 0);
    }

    // --- Stage 5: pooled = segment_sum(x2, idx) ---
    zero_kernel<<<gsBlocks((long long)N_GRAPHS * H2), 256, 0, stream>>>(
        pooled, (long long)N_GRAPHS * H2);
    pool_kernel<<<gsBlocks((long long)N_NODES * H2), 256, 0, stream>>>(
        R1, idx, pooled, (long long)N_NODES * H2, 8 /*log2(256)*/);

    // --- Stage 6: head + log_softmax ---
    head_kernel<<<N_GRAPHS, H2, 0, stream>>>(pooled, W3, b3, W4, b4, out);
}
